// ContrastiveLoss_37701222924996
// MI455X (gfx1250) — matrix operand order repaired and checked by reading, not compiled, not run on hardware
//
#include <hip/hip_runtime.h>
#include <hip/hip_bf16.h>

typedef __attribute__((ext_vector_type(16))) _Float16 v16h;
typedef __attribute__((ext_vector_type(8)))  _Float16 h8;
typedef __attribute__((ext_vector_type(4)))  _Float16 h4;
typedef __attribute__((ext_vector_type(8)))  float    v8f;

#define BROWS 4096
#define DIM   512
#define N2    8192
// exp(s/0.5) = exp2(s * 2/ln2)
#define INV_T_LOG2E 2.8853900817779268f

// ---------------- init: labels (int32) + zero row sums --------------------
__global__ void k_init(const int* __restrict__ target, int* __restrict__ lab,
                       float* __restrict__ rowsum) {
    int i = blockIdx.x * blockDim.x + threadIdx.x;
    if (i < N2) {
        lab[i]    = target[i & (BROWS - 1)];
        rowsum[i] = 0.0f;
    }
}

// ---------------- L2 normalize rows, f32 -> f16 ---------------------------
__global__ void k_norm(const float* __restrict__ emb_i,
                       const float* __restrict__ emb_j,
                       _Float16* __restrict__ z) {
    int wave = (blockIdx.x * blockDim.x + threadIdx.x) >> 5;
    int lane = threadIdx.x & 31;
    if (wave >= N2) return;
    const float* src = (wave < BROWS) ? (emb_i + (size_t)wave * DIM)
                                      : (emb_j + (size_t)(wave - BROWS) * DIM);
    float4 v[4];
    float ss = 0.0f;
    #pragma unroll
    for (int q = 0; q < 4; ++q) {
        v[q] = reinterpret_cast<const float4*>(src)[q * 32 + lane];
        ss += v[q].x * v[q].x + v[q].y * v[q].y + v[q].z * v[q].z + v[q].w * v[q].w;
    }
    #pragma unroll
    for (int off = 16; off >= 1; off >>= 1) ss += __shfl_xor(ss, off, 32);
    float inv = 1.0f / fmaxf(sqrtf(ss), 1e-12f);
    _Float16* dst = z + (size_t)wave * DIM;
    #pragma unroll
    for (int q = 0; q < 4; ++q) {
        h4 o;
        o[0] = (_Float16)(v[q].x * inv);
        o[1] = (_Float16)(v[q].y * inv);
        o[2] = (_Float16)(v[q].z * inv);
        o[3] = (_Float16)(v[q].w * inv);
        reinterpret_cast<h4*>(dst)[q * 32 + lane] = o;
    }
}

// ---------------- positives: pos[r] = z[r] . z[r+B] -----------------------
__global__ void k_pos(const _Float16* __restrict__ z, float* __restrict__ pos) {
    int wave = (blockIdx.x * blockDim.x + threadIdx.x) >> 5;
    int lane = threadIdx.x & 31;
    if (wave >= BROWS) return;
    const h8* a = reinterpret_cast<const h8*>(z + (size_t)wave * DIM);
    const h8* b = reinterpret_cast<const h8*>(z + (size_t)(wave + BROWS) * DIM);
    float s = 0.0f;
    #pragma unroll
    for (int q = 0; q < 2; ++q) {
        h8 x = a[q * 32 + lane];
        h8 y = b[q * 32 + lane];
        #pragma unroll
        for (int e = 0; e < 8; ++e) s += (float)x[e] * (float)y[e];
    }
    #pragma unroll
    for (int off = 16; off >= 1; off >>= 1) s += __shfl_xor(s, off, 32);
    if (lane == 0) pos[wave] = s;
}

// ---------------- fused WMMA GEMM + mask + exp + row-sum ------------------
// Block: 256 threads = 8 waves. Each block owns one 16-row strip (A in LDS).
// Each wave computes a 16x64 tile of sim with a software-pipelined
// (2-deep ping-pong) K loop: step k+1's A/B fragments load while step k's
// four WMMAs execute, keeping LOADcnt waits partial instead of full drains.
// Mask/exp/row-reduction fused so sim is never materialized.
__global__ void __launch_bounds__(256) k_main(const _Float16* __restrict__ z,
                                              const int* __restrict__ lab,
                                              float* __restrict__ rowsum) {
    __shared__ _Float16 Alds[16 * DIM];   // 16 KB A strip
    const int ib = blockIdx.x * 16;

    // cooperative stage of the 16x512 A strip into LDS
    {
        const int4* src = reinterpret_cast<const int4*>(z + (size_t)ib * DIM);
        int4* dst = reinterpret_cast<int4*>(Alds);
        #pragma unroll
        for (int q = 0; q < 4; ++q)
            dst[threadIdx.x + 256 * q] = src[threadIdx.x + 256 * q];
    }
    __syncthreads();

    const int waveId = threadIdx.x >> 5;
    const int lane   = threadIdx.x & 31;
    const int n      = lane & 15;     // column within tile / A row
    const int h      = lane >> 4;     // lane half
    const int jb     = (blockIdx.y * 8 + waveId) * 64;

    // A fragment source in LDS (per-lane base; kb applied as offset)
    const _Float16* abase0 = Alds + (size_t)n * DIM + 8 * h;
    // B fragment bases in global (per-lane; kb applied as offset)
    const _Float16* bbase[4];
    #pragma unroll
    for (int t = 0; t < 4; ++t)
        bbase[t] = z + (size_t)(jb + t * 16 + n) * DIM + 8 * h;

    const v8f vzero = {0.f, 0.f, 0.f, 0.f, 0.f, 0.f, 0.f, 0.f};
    v8f acc[4];
    #pragma unroll
    for (int t = 0; t < 4; ++t) acc[t] = vzero;

    // ping-pong fragment buffers
    v16h abuf[2];
    v16h bbuf[2][4];

    // prologue: fill buffer 0 with K-step 0
    {
        int4* ap = reinterpret_cast<int4*>(&abuf[0]);
        ap[0] = *reinterpret_cast<const int4*>(abase0);
        ap[1] = *reinterpret_cast<const int4*>(abase0 + 16);
        #pragma unroll
        for (int t = 0; t < 4; ++t) {
            int4* bp = reinterpret_cast<int4*>(&bbuf[0][t]);
            bp[0] = reinterpret_cast<const int4*>(bbase[t])[0];
            bp[1] = reinterpret_cast<const int4*>(bbase[t])[2];
        }
    }

    #pragma unroll
    for (int step = 0; step < DIM / 32; ++step) {
        const int cur = step & 1;
        const int nxt = cur ^ 1;
        const int kn  = (step + 1) * 32;
        // issue next step's loads BEFORE consuming current fragments
        if (kn < DIM) {
            int4* ap = reinterpret_cast<int4*>(&abuf[nxt]);
            ap[0] = *reinterpret_cast<const int4*>(abase0 + kn);
            ap[1] = *reinterpret_cast<const int4*>(abase0 + kn + 16);
            #pragma unroll
            for (int t = 0; t < 4; ++t) {
                int4* bp = reinterpret_cast<int4*>(&bbuf[nxt][t]);
                bp[0] = reinterpret_cast<const int4*>(bbase[t] + kn)[0];
                bp[1] = reinterpret_cast<const int4*>(bbase[t] + kn)[2];
            }
        }
        #pragma unroll
        for (int t = 0; t < 4; ++t)
            acc[t] = __builtin_amdgcn_wmma_f32_16x16x32_f16(
                false, abuf[cur], false, bbuf[cur][t], (short)0, acc[t],
                false, false);
    }

    // epilogue: mask, exp(sim/tau), per-row partial sums, atomic accumulate.
    // C layout: element (M = r + 8h, N = n) in acc[t][r].
    int labj[4];
    #pragma unroll
    for (int t = 0; t < 4; ++t) labj[t] = lab[jb + t * 16 + n];

    #pragma unroll
    for (int r = 0; r < 8; ++r) {
        const int i   = ib + r + 8 * h;
        const int li  = lab[i];
        float rp = 0.0f;
        #pragma unroll
        for (int t = 0; t < 4; ++t) {
            const int j = jb + t * 16 + n;
            const int d = i - j;
            // keep iff j != i AND (different label OR positive pair |i-j|==B)
            const bool inc = (d != 0) && ((li != labj[t]) || (d == BROWS) || (d == -BROWS));
            const float e = exp2f(acc[t][r] * INV_T_LOG2E);
            rp += inc ? e : 0.0f;
        }
        // reduce over the 16 lanes holding row i (width-16 xor shuffle)
        #pragma unroll
        for (int off = 8; off >= 1; off >>= 1) rp += __shfl_xor(rp, off, 16);
        if (n == 0) atomicAdd(&rowsum[i], rp);
    }
}

// ---------------- finalize: loss = mean(log(denom+1e-7) - pos/tau) --------
__global__ void __launch_bounds__(1024) k_fin(const float* __restrict__ rowsum,
                                              const float* __restrict__ pos,
                                              float* __restrict__ out) {
    __shared__ float red[1024];
    float s = 0.0f;
    #pragma unroll
    for (int q = 0; q < 8; ++q) {
        int i = threadIdx.x + q * 1024;
        float p = pos[i & (BROWS - 1)];          // row i and i+B share the positive
        s += logf(rowsum[i] + 1e-7f) - 2.0f * p; // pos / tau, tau = 0.5
    }
    red[threadIdx.x] = s;
    __syncthreads();
    for (int st = 512; st > 0; st >>= 1) {
        if (threadIdx.x < st) red[threadIdx.x] += red[threadIdx.x + st];
        __syncthreads();
    }
    if (threadIdx.x == 0) out[0] = red[0] / (float)N2;
}

extern "C" void kernel_launch(void* const* d_in, const int* in_sizes, int n_in,
                              void* d_out, int out_size, void* d_ws, size_t ws_size,
                              hipStream_t stream) {
    const float* emb_i  = (const float*)d_in[0];
    const float* emb_j  = (const float*)d_in[1];
    const int*   target = (const int*)d_in[2];
    float* out = (float*)d_out;

    // workspace layout
    char* ws = (char*)d_ws;
    _Float16* z     = (_Float16*)ws;                          // 8192*512*2 = 8 MiB
    float* rowsum   = (float*)(ws + (size_t)N2 * DIM * 2);    // 32 KiB
    float* pos      = rowsum + N2;                            // 16 KiB
    int*   lab      = (int*)(pos + BROWS);                    // 32 KiB

    k_init<<<(N2 + 255) / 256, 256, 0, stream>>>(target, lab, rowsum);
    k_norm<<<(N2 * 32) / 256, 256, 0, stream>>>(emb_i, emb_j, z);
    k_pos<<<(BROWS * 32) / 256, 256, 0, stream>>>(z, pos);
    k_main<<<dim3(N2 / 16, N2 / 512), 256, 0, stream>>>(z, lab, rowsum);
    k_fin<<<1, 1024, 0, stream>>>(rowsum, pos, out);
}
